// NonZeroFeatureExtractor_53060025975188
// MI455X (gfx1250) — matrix operand order repaired and checked by reading, not compiled, not run on hardware
//
#include <hip/hip_runtime.h>

// CDNA5 / gfx1250 WMMA vector types
typedef __attribute__((ext_vector_type(16))) _Float16 v16h;
typedef __attribute__((ext_vector_type(8)))  float    v8f;

static constexpr int IMG_H = 1024;
static constexpr int IMG_W = 1024;
static constexpr int BATCH = 8;
static constexpr int TW    = 64;          // output tile width
static constexpr int TH    = 32;          // output tile height
static constexpr int HALO  = 16;          // max s/2
static constexpr int RC    = TW + 2*HALO; // 96 region cols
static constexpr int RR    = TH + 2*HALO; // 64 region rows
static constexpr int RSTR  = 100;         // lumS stride: mult of 4 -> 16B-aligned rows
static constexpr int VW    = 112;         // vsum width incl. zero pad (max idx 111)
static constexpr int VSTR  = 113;         // odd stride: conflict-free column access

__launch_bounds__(256)
__global__ void nzfeat_kernel(const float* __restrict__ x, float* __restrict__ out)
{
    __shared__ float lumS[RR * RSTR];     // 64x96 lum with halo (25.6 KB)
    __shared__ float vsumL[TH * VSTR];    // vertical box sums (lum)   (14.5 KB)
    __shared__ float vsumC[TH * VSTR];    // vertical box sums (count) (14.5 KB)

    const int tid = threadIdx.x;
    const int b   = blockIdx.z;
    const int tr0 = blockIdx.y * TH;
    const int tc0 = blockIdx.x * TW;

    const float* x0 = x + (size_t)b * 3 * IMG_H * IMG_W;
    const float* x1 = x0 + (size_t)IMG_H * IMG_W;
    const float* x2 = x1 + (size_t)IMG_H * IMG_W;

    // ---- Zero-fill vsum pad columns [RC, VW) once: out-of-band WMMA A reads hit 0 ----
    for (int idx = tid; idx < TH * (VW - RC) * 2; idx += 256) {
        const int half = TH * (VW - RC);
        float* dst = (idx >= half) ? vsumC : vsumL;
        const int rem = (idx >= half) ? idx - half : idx;
        dst[(rem >> 4) * VSTR + RC + (rem & 15)] = 0.0f;
    }

    // ---- Phase 1: float4 halo loads, lum = channel mean (zero padding at edges) ----
    // tile origins are multiples of 64, halo 16 -> every 4-wide load is aligned and
    // either fully inside or fully outside the image (no straddles).
    for (int j = tid; j < RR * (RC / 4); j += 256) {
        const int rr  = j / (RC / 4);
        const int cc4 = (j - rr * (RC / 4)) * 4;
        const int gr  = tr0 - HALO + rr;
        const int gc  = tc0 - HALO + cc4;
        float4 l = make_float4(0.f, 0.f, 0.f, 0.f);
        if (gr >= 0 && gr < IMG_H && gc >= 0 && gc + 3 < IMG_W) {
            const size_t o = (size_t)gr * IMG_W + gc;
            const float4 a = *(const float4*)(x0 + o);
            const float4 c1 = *(const float4*)(x1 + o);
            const float4 c2 = *(const float4*)(x2 + o);
            l.x = (a.x + c1.x + c2.x) * (1.0f / 3.0f);
            l.y = (a.y + c1.y + c2.y) * (1.0f / 3.0f);
            l.z = (a.z + c1.z + c2.z) * (1.0f / 3.0f);
            l.w = (a.w + c1.w + c2.w) * (1.0f / 3.0f);
        }
        *(float4*)&lumS[rr * RSTR + cc4] = l;   // (rr*100 + cc4)*4 is 16B aligned
    }
    __syncthreads();

    const int lane = tid & 31;
    const int wv   = tid >> 5;            // 8 waves -> 2x4 grid of 16x16 tiles
    const int wtr  = (wv >> 2) * 16;      // wave tile row offset (0 / 16)
    const int wtc  = (wv & 3)  * 16;      // wave tile col offset (0/16/32/48)
    const int arow = lane & 15;           // matrix row (A) / column (B/D) of this lane
    const int kgrp = lane >> 4;           // half-wave group

    const int scales[4] = {3, 9, 17, 33};

    for (int si = 0; si < 4; ++si) {
        const int s = scales[si];
        const int R = s >> 1;
        const int span = 16 + 2 * R;      // K extent of banded matmul (<= 48)

        // ---- Phase 2: vertical running box sums. Jobs: tid<96 lum cols, 96..191 cnt
        // cols (mask recomputed from lum: exact 0/1 integers, f32 running sum exact).
        if (tid < 2 * RC) {
            const bool isCnt = tid >= RC;
            const int  c = isCnt ? tid - RC : tid;
            float* dst = isCnt ? vsumC : vsumL;
            float acc = 0.0f;
            for (int k = 0; k <= 2 * R; ++k) {
                float v = lumS[(HALO - R + k) * RSTR + c];
                acc += isCnt ? ((v != 0.0f) ? 1.0f : 0.0f) : v;
            }
            dst[c] = acc;
            for (int r = 1; r < TH; ++r) {
                float va = lumS[(HALO + R + r) * RSTR + c];
                float vs = lumS[(HALO - R + r - 1) * RSTR + c];
                if (isCnt) {
                    va = (va != 0.0f) ? 1.0f : 0.0f;
                    vs = (vs != 0.0f) ? 1.0f : 0.0f;
                }
                acc += va - vs;
                dst[r * VSTR + c] = acc;
            }
        }
        __syncthreads();

        // ---- Phase 3: horizontal box sums as banded matmuls on the WMMA pipe ----
        // out(16x16) = Vsum(16xK) x Band(KxN),  Band[k][n] = (n <= k <= n+2R)
        // A 16x32 f16 lane layout (ISA 7.12.2): K = h + 8*kgrp + (h & 8), row = lane&15
        // B 32x16 f16 lane layout:              K = h + 8*kgrp + (h & 8), col = lane&15
        v8f accL = {};   // lum sums: 2-term f16 split, f32 accum -> ~f32 accurate
        v8f accC = {};   // counts: exact (f16 ints <= 33, 0/1 band, f32 accum)
        const int nchunk = (span > 32) ? 2 : 1;
        const int cbase  = wtc + HALO - R;        // vsum col of k=0 (>= 0)
        for (int q = 0; q < nchunk; ++q) {
            v16h aHi, aLo, aC, bB;
#pragma unroll
            for (int h = 0; h < 16; ++h) {
                const int kA  = 32 * q + h + 8 * kgrp + (h & 8);
                const int col = cbase + kA;               // <= 111, pad cols are zero
                const float vl = vsumL[(wtr + arow) * VSTR + col];
                const _Float16 hi = (_Float16)vl;
                aHi[h] = hi;
                aLo[h] = (_Float16)(vl - (float)hi);
                aC[h]  = (_Float16)vsumC[(wtr + arow) * VSTR + col];
                const int kB = 32 * q + h + 8 * kgrp + (h & 8);
                bB[h] = (kB >= arow && kB <= arow + 2 * R) ? (_Float16)1.0f
                                                           : (_Float16)0.0f;
            }
            accL = __builtin_amdgcn_wmma_f32_16x16x32_f16(false, aHi, false, bB,
                                                          (short)0, accL, false, false);
            accL = __builtin_amdgcn_wmma_f32_16x16x32_f16(false, aLo, false, bB,
                                                          (short)0, accL, false, false);
            accC = __builtin_amdgcn_wmma_f32_16x16x32_f16(false, aC,  false, bB,
                                                          (short)0, accC, false, false);
        }

        // ---- Phase 4: write the two feature channels for this scale ----
        const float inv_s2 = 1.0f / (float)(s * s);
        float* outP = out + ((size_t)b * 8 + 2 * si) * IMG_H * IMG_W; // count fraction
        float* outQ = outP + (size_t)IMG_H * IMG_W;                   // nonzero-mean pool
#pragma unroll
        for (int i = 0; i < 8; ++i) {
            const int gr = tr0 + wtr + i + 8 * kgrp;   // C/D layout: M = i + 8*(lane>>4)
            const int gc = tc0 + wtc + arow;           //             N = lane & 15
            const size_t o = (size_t)gr * IMG_W + gc;
            const float cnt = accC[i];
            outP[o] = cnt * inv_s2;
            outQ[o] = accL[i] / fmaxf(cnt, 1.0f);
        }
        __syncthreads();   // vsum buffers reused next scale
    }
}

extern "C" void kernel_launch(void* const* d_in, const int* in_sizes, int n_in,
                              void* d_out, int out_size, void* d_ws, size_t ws_size,
                              hipStream_t stream)
{
    const float* x  = (const float*)d_in[0];
    float* out      = (float*)d_out;
    dim3 grid(IMG_W / TW, IMG_H / TH, BATCH);   // 16 x 32 x 8 workgroups
    nzfeat_kernel<<<grid, 256, 0, stream>>>(x, out);
}
